// SPN_36936718745981
// MI455X (gfx1250) — matrix operand order repaired and checked by reading, not compiled, not run on hardware
//
#include <hip/hip_runtime.h>
#include <math.h>

typedef __attribute__((ext_vector_type(16))) _Float16 v16h;
typedef __attribute__((ext_vector_type(8)))  _Float16 v8h;
typedef __attribute__((ext_vector_type(8)))  float    v8f;

#define DEV __device__ __forceinline__

namespace spn {

constexpr int H  = 256, W  = 1216, P  = H * W;    // full res
constexpr int H2 = 128, W2 = 608,  P2 = H2 * W2;  // half res

union AU { v16h v; v8h h[2]; };

// ------------------------------------------------------------------
// Weight prep: pack fused [mask|kconf|g3|g5|g7] 3x3 weights (Cout=84,
// padded to 96) into per-lane WMMA B-fragment order, fp16.
// Layout: wp[((s*6+n)*32 + lane)*16 + i] = W[K=s*32+(i<8 ? 8*hi+i : 16+8*hi+i-8)][oc=n*16+m]
// K order = tap*CIN + c  (tap = 3x3 position, row major)
// ------------------------------------------------------------------
__global__ __launch_bounds__(256) void prep_w_kernel(
    _Float16* __restrict__ wp, int CIN, int KSTEPS,
    const float* __restrict__ w_m, const float* __restrict__ w_k,
    const float* __restrict__ w_3, const float* __restrict__ w_5,
    const float* __restrict__ w_7)
{
  int id = blockIdx.x * 256 + threadIdx.x;
  int total = KSTEPS * 6 * 32 * 16;
  if (id >= total) return;
  int i    = id & 15;
  int lane = (id >> 4) & 31;
  int sn   = id >> 9;
  int s = sn / 6, n = sn % 6;
  int hi = lane >> 4, m = lane & 15;
  int oc = n * 16 + m;
  int K  = s * 32 + ((i < 8) ? 8 * hi + i : 16 + 8 * hi + (i - 8));
  int tap = K / CIN, c = K % CIN;
  float val = 0.f;
  if      (oc < 1)  val = w_m[(oc * CIN + c) * 9 + tap];
  else if (oc < 4)  val = w_k[((oc - 1)  * CIN + c) * 9 + tap];
  else if (oc < 12) val = w_3[((oc - 4)  * CIN + c) * 9 + tap];
  else if (oc < 36) val = w_5[((oc - 12) * CIN + c) * 9 + tap];
  else if (oc < 84) val = w_7[((oc - 36) * CIN + c) * 9 + tap];
  wp[id] = (_Float16)val;
}

// Fold BN into per-channel (scale, bias): out = acc*s + t
__global__ __launch_bounds__(128) void prep_sb_kernel(float2* __restrict__ sb,
  const float* bm, const float* gm, const float* bem, const float* mm, const float* vm,
  const float* bk, const float* gk, const float* bek, const float* mk, const float* vk,
  const float* b3, const float* g3, const float* be3, const float* m3, const float* v3,
  const float* b5, const float* g5, const float* be5, const float* m5, const float* v5,
  const float* b7, const float* g7, const float* be7, const float* m7, const float* v7)
{
  int c = threadIdx.x;
  if (c >= 96) return;
  const float *b, *g, *be, *mu, *v; int l;
  if      (c < 1)  { b=bm; g=gm; be=bem; mu=mm; v=vm; l=c;      }
  else if (c < 4)  { b=bk; g=gk; be=bek; mu=mk; v=vk; l=c-1;    }
  else if (c < 12) { b=b3; g=g3; be=be3; mu=m3; v=v3; l=c-4;    }
  else if (c < 36) { b=b5; g=g5; be=be5; mu=m5; v=v5; l=c-12;   }
  else if (c < 84) { b=b7; g=g7; be=be7; mu=m7; v=v7; l=c-36;   }
  else { sb[c] = make_float2(0.f, 0.f); return; }
  float s = g[l] * rsqrtf(v[l] + 1e-5f);
  sb[c] = make_float2(s, (b[l] - mu[l]) * s + be[l]);
}

// ------------------------------------------------------------------
// Outlier filter: separable 7x7 zero-pad box sums over sparse & mask,
// then the full inlier logic fused in the vertical pass.
// ------------------------------------------------------------------
__global__ __launch_bounds__(256) void box_h_kernel(const float* __restrict__ sp,
                                                    float2* __restrict__ tmp)
{
  int pix = blockIdx.x * 256 + threadIdx.x;
  if (pix >= P) return;
  int y = pix / W, x = pix - y * W;
  float ss = 0.f, ms = 0.f;
  #pragma unroll
  for (int dx = -3; dx <= 3; ++dx) {
    int xx = x + dx;
    if (xx >= 0 && xx < W) {
      float v = sp[y * W + xx];
      ss += v;
      ms += (v > 0.f) ? 1.f : 0.f;
    }
  }
  tmp[pix] = make_float2(ss, ms);
}

__global__ __launch_bounds__(256) void box_v_filter_kernel(
    const float* __restrict__ sp, const float* __restrict__ depth,
    const float2* __restrict__ tmp, const float* __restrict__ fbias_p,
    float* __restrict__ sparseF, float* __restrict__ validF)
{
  int pix = blockIdx.x * 256 + threadIdx.x;
  if (pix >= P) return;
  int y = pix / W, x = pix - y * W;
  float ss = 0.f, ms = 0.f;
  #pragma unroll
  for (int dy = -3; dy <= 3; ++dy) {
    int yy = y + dy;
    if (yy >= 0 && yy < H) {
      float2 t = tmp[yy * W + x];
      ss += t.x; ms += t.y;
    }
  }
  float fb = fbias_p[0];
  float s0 = sp[pix];
  float mk = (s0 > 0.f) ? 1.f : 0.f;
  float avg = (ss + fb) / ((ms + fb) + 1e-5f) * mk;
  float inl = ((s0 - avg) < 1.f) ? 1.f : 0.f;
  float s1 = s0 * inl;
  float m0 = (s1 > 0.f) ? 1.f : 0.f;
  float resid = (s1 - depth[pix]) * m0;
  float inl2 = (resid < 1.f) ? 1.f : 0.f;
  float sf = s1 * inl2;
  sparseF[pix] = sf;
  validF[pix]  = (sf > 0.f) ? 1.f : 0.f;
}

// sparse_downsample_close (2x2 "closest valid" min-pool)
__global__ __launch_bounds__(256) void downsample_kernel(
    const float* __restrict__ sf, const float* __restrict__ vm,
    float* __restrict__ ds2, float* __restrict__ vms2)
{
  int pix2 = blockIdx.x * 256 + threadIdx.x;
  if (pix2 >= P2) return;
  int y = pix2 / W2, x = pix2 - y * W2;
  const float L = 600.f;
  float e = -3.402823e38f, m = 0.f;
  #pragma unroll
  for (int dy = 0; dy < 2; ++dy)
    #pragma unroll
    for (int dx = 0; dx < 2; ++dx) {
      int idx = (2 * y + dy) * W + (2 * x + dx);
      float mk = vm[idx];
      float enc = -(1.f - mk) * L - sf[idx];
      e = fmaxf(e, enc);
      m = fmaxf(m, mk);
    }
  ds2[pix2]  = (-e) - (1.f - m) * L;
  vms2[pix2] = m;
}

// ------------------------------------------------------------------
// 1x1 proj conv + BN, output pixel-major fp16 (feeds WMMA A-operand)
// ------------------------------------------------------------------
template <int CIN>
__global__ __launch_bounds__(256) void proj_kernel(
    const float* __restrict__ fin, const float* __restrict__ wgt,
    const float* __restrict__ bb, const float* __restrict__ gg,
    const float* __restrict__ bbe, const float* __restrict__ mm,
    const float* __restrict__ vv, _Float16* __restrict__ outp, int NP)
{
  __shared__ float ws[CIN * CIN];
  __shared__ float sS[CIN], sT[CIN];
  int tid = threadIdx.x;
  for (int i = tid; i < CIN * CIN; i += 256) ws[i] = wgt[i];
  if (tid < CIN) {
    float s = gg[tid] * rsqrtf(vv[tid] + 1e-5f);
    sS[tid] = s;
    sT[tid] = (bb[tid] - mm[tid]) * s + bbe[tid];
  }
  __syncthreads();
  int pix = blockIdx.x * 256 + tid;
  if (pix >= NP) return;
  float xv[CIN];
  #pragma unroll
  for (int c = 0; c < CIN; ++c) xv[c] = fin[(size_t)c * NP + pix];
  for (int oc = 0; oc < CIN; ++oc) {
    float a = 0.f;
    #pragma unroll
    for (int c = 0; c < CIN; ++c) a = fmaf(ws[oc * CIN + c], xv[c], a);
    outp[(size_t)pix * CIN + oc] = (_Float16)(a * sS[oc] + sT[oc]);
  }
}

// Guide normalization epilogue helper: L1-normalize, insert mid channel,
// write k*k fp16 planes.
template <int NC>
DEV void writeGuide(const float* raw, _Float16* __restrict__ out, int g, int NP)
{
  float S = 0.f;
  #pragma unroll
  for (int i = 0; i < NC; ++i) S += fabsf(raw[i]);
  float inv = 1.f / S;
  float sum = 0.f;
  constexpr int HALF = NC / 2;
  #pragma unroll
  for (int i = 0; i < NC; ++i) {
    float gn = raw[i] * inv;
    sum += gn;
    int j = (i < HALF) ? i : i + 1;
    out[(size_t)j * NP + g] = (_Float16)gn;
  }
  out[(size_t)HALF * NP + g] = (_Float16)(1.f - sum);
}

// ------------------------------------------------------------------
// Fused 3x3 conv heads via WMMA implicit GEMM.
// Block = 64 pixels x 96 channels. 8 waves: wave = (mt<<1)|nh, each wave
// owns 1 M-tile x 3 N-tiles. Epilogue (via LDS): BN fold, sigmoid mask,
// softmax kconf, L1 guide normalization + mid channel.
// ------------------------------------------------------------------
template <int CIN, int HH, int WW>
__global__ __launch_bounds__(256) void conv3_gemm_kernel(
    const _Float16* __restrict__ xin,   // [NP][CIN] pixel-major fp16
    const _Float16* __restrict__ wpack, // prepacked B fragments
    const float2*  __restrict__ sb,     // folded BN scale/bias [96]
    const float*   __restrict__ validMask,
    float* __restrict__ maskOut, float* __restrict__ kcOut,
    _Float16* __restrict__ g3o, _Float16* __restrict__ g5o,
    _Float16* __restrict__ g7o)
{
  constexpr int NP = HH * WW;
  constexpr int KSTEPS = CIN * 9 / 32;
  __shared__ float sTile[64 * 96];

  const int tid  = threadIdx.x;
  const int lane = tid & 31, wave = tid >> 5;
  const int mt = wave >> 1, nh = wave & 1;
  const int m  = lane & 15, hi = lane >> 4;
  const int pixBase = blockIdx.x * 64;
  const int pix = pixBase + mt * 16 + m;
  const int y = pix / WW, x = pix - y * WW;

  v8f acc0 = {}, acc1 = {}, acc2 = {};

  for (int s = 0; s < KSTEPS; ++s) {
    int tap, coff;
    if (CIN == 32) { tap = s; coff = 0; }
    else           { tap = s >> 1; coff = (s & 1) * 32; }
    int dy = tap / 3 - 1;
    int dx = tap - (tap / 3) * 3 - 1;
    int yn = y + dy; yn = (yn < 0) ? -yn : ((yn >= HH) ? 2 * HH - 2 - yn : yn);
    int xn = x + dx; xn = (xn < 0) ? -xn : ((xn >= WW) ? 2 * WW - 2 - xn : xn);
    // A fragment: lane holds row M=m; K slots = {8hi+0..7, 16+8hi+0..7}
    const _Float16* ap = xin + (size_t)(yn * WW + xn) * CIN + coff + hi * 8;
    AU a;
    a.h[0] = *(const v8h*)(ap);
    a.h[1] = *(const v8h*)(ap + 16);
    // B fragments: 16 contiguous halfs per lane, prepacked
    const _Float16* wp = wpack + ((size_t)(s * 6 + nh * 3) * 32 + lane) * 16;
    AU b0, b1, b2;
    b0.h[0] = *(const v8h*)(wp);        b0.h[1] = *(const v8h*)(wp + 8);
    b1.h[0] = *(const v8h*)(wp + 512);  b1.h[1] = *(const v8h*)(wp + 520);
    b2.h[0] = *(const v8h*)(wp + 1024); b2.h[1] = *(const v8h*)(wp + 1032);
    acc0 = __builtin_amdgcn_wmma_f32_16x16x32_f16(false, a.v, false, b0.v, (short)0, acc0, false, false);
    acc1 = __builtin_amdgcn_wmma_f32_16x16x32_f16(false, a.v, false, b1.v, (short)0, acc1, false, false);
    acc2 = __builtin_amdgcn_wmma_f32_16x16x32_f16(false, a.v, false, b2.v, (short)0, acc2, false, false);
  }

  // BN fold + stage to LDS (D layout: lane column N=m, rows M=r+8*hi)
  #pragma unroll
  for (int q = 0; q < 3; ++q) {
    v8f A = (q == 0) ? acc0 : ((q == 1) ? acc1 : acc2);
    int ch = (nh * 3 + q) * 16 + m;
    float2 st = sb[ch];
    #pragma unroll
    for (int r = 0; r < 8; ++r) {
      int pl = mt * 16 + r + 8 * hi;
      sTile[pl * 96 + ch] = A[r] * st.x + st.y;
    }
  }
  __syncthreads();

  if (tid < 64) {
    int g = pixBase + tid;
    const float* row = sTile + tid * 96;
    // mask head: sigmoid * valid
    maskOut[g] = (1.f / (1.f + expf(-row[0]))) * validMask[g];
    // kconf head: softmax over 3
    float a0 = row[1], a1 = row[2], a2 = row[3];
    float mx = fmaxf(a0, fmaxf(a1, a2));
    float e0 = expf(a0 - mx), e1 = expf(a1 - mx), e2 = expf(a2 - mx);
    float es = e0 + e1 + e2;
    kcOut[(size_t)0 * NP + g] = e0 / es;
    kcOut[(size_t)1 * NP + g] = e1 / es;
    kcOut[(size_t)2 * NP + g] = e2 / es;
    // guide heads: L1 normalize + mid channel
    writeGuide<8>(row + 4,  g3o, g, NP);
    writeGuide<24>(row + 12, g5o, g, NP);
    writeGuide<48>(row + 36, g7o, g, NP);
  }
}

// ------------------------------------------------------------------
// CSPN step. kernel_trans + nnup folded into the gather:
//   weight tap t  = guide[k^2-1-t] @ ((y>>GSH)+rt-c, (x>>GSH)+ct-c)  (zero OOB)
//   value  tap t  = din @ (y+(rt-c)*DIL, x+(ct-c)*DIL)               (zero OOB)
//   mid tap       = guide[mid]@center * d0
// ------------------------------------------------------------------
template <int K, int DIL, int GSH>
DEV float cspn_acc(const _Float16* __restrict__ g, const float* __restrict__ din,
                   float d0, int y, int x, int Hg, int Wg)
{
  constexpr int C = (K - 1) / 2, K2 = K * K, MID = (K2 - 1) / 2;
  const int yg = y >> GSH, xg = x >> GSH;
  const int NPg = Hg * Wg;
  float acc = 0.f;
  #pragma unroll
  for (int rt = 0; rt < K; ++rt) {
    #pragma unroll
    for (int ct = 0; ct < K; ++ct) {
      int t = rt * K + ct;
      if (t == MID) {
        acc += (float)g[(size_t)MID * NPg + yg * Wg + xg] * d0;
      } else {
        int j = K2 - 1 - t;
        int ys = yg + rt - C, xs = xg + ct - C;
        float w = (ys >= 0 && ys < Hg && xs >= 0 && xs < Wg)
                      ? (float)g[(size_t)j * NPg + ys * Wg + xs] : 0.f;
        int yn = y + (rt - C) * DIL, xn = x + (ct - C) * DIL;
        float v = (yn >= 0 && yn < H && xn >= 0 && xn < W) ? din[yn * W + xn] : 0.f;
        acc += w * v;
      }
    }
  }
  return acc;
}

__global__ __launch_bounds__(256) void cspn_s2_kernel(
    const float* __restrict__ d3i, const float* __restrict__ d5i,
    const float* __restrict__ d7i, float* __restrict__ d3o,
    float* __restrict__ d5o, float* __restrict__ d7o,
    const _Float16* __restrict__ g3, const _Float16* __restrict__ g5,
    const _Float16* __restrict__ g7, const float* __restrict__ depth0,
    const float* __restrict__ maskS2, const float* __restrict__ ds2)
{
  int pix = blockIdx.x * 256 + threadIdx.x;
  if (pix >= P) return;
  int y = pix / W, x = pix - y * W;
  int hidx = (y >> 1) * W2 + (x >> 1);
  float mv = maskS2[hidx];
  float blend = mv * ds2[hidx];
  float om = 1.f - mv;
  float d0 = depth0[pix];
  d3o[pix] = blend + om * cspn_acc<3, 2, 1>(g3, d3i, d0, y, x, H2, W2);
  d5o[pix] = blend + om * cspn_acc<5, 2, 1>(g5, d5i, d0, y, x, H2, W2);
  d7o[pix] = blend + om * cspn_acc<7, 2, 1>(g7, d7i, d0, y, x, H2, W2);
}

__global__ __launch_bounds__(256) void cspn_full_kernel(
    const float* __restrict__ d3i, const float* __restrict__ d5i,
    const float* __restrict__ d7i, float* __restrict__ d3o,
    float* __restrict__ d5o, float* __restrict__ d7o,
    const _Float16* __restrict__ g3, const _Float16* __restrict__ g5,
    const _Float16* __restrict__ g7, const float* __restrict__ refined,
    const float* __restrict__ maskF, const float* __restrict__ sparseF)
{
  int pix = blockIdx.x * 256 + threadIdx.x;
  if (pix >= P) return;
  int y = pix / W, x = pix - y * W;
  float mv = maskF[pix];
  float blend = mv * sparseF[pix];
  float om = 1.f - mv;
  float d0 = refined[pix];
  d3o[pix] = blend + om * cspn_acc<3, 1, 0>(g3, d3i, d0, y, x, H, W);
  d5o[pix] = blend + om * cspn_acc<5, 1, 0>(g5, d5i, d0, y, x, H, W);
  d7o[pix] = blend + om * cspn_acc<7, 1, 0>(g7, d7i, d0, y, x, H, W);
}

__global__ __launch_bounds__(256) void blend_s2_kernel(
    const float* __restrict__ d3, const float* __restrict__ d5,
    const float* __restrict__ d7, const float* __restrict__ kcS,
    float* __restrict__ refined)
{
  int pix = blockIdx.x * 256 + threadIdx.x;
  if (pix >= P) return;
  int y = pix / W, x = pix - y * W;
  int hidx = (y >> 1) * W2 + (x >> 1);
  refined[pix] = kcS[hidx] * d3[pix] + kcS[(size_t)P2 + hidx] * d5[pix] +
                 kcS[(size_t)2 * P2 + hidx] * d7[pix];
}

__global__ __launch_bounds__(256) void blend_final_kernel(
    const float* __restrict__ d3, const float* __restrict__ d5,
    const float* __restrict__ d7, const float* __restrict__ kcF,
    float* __restrict__ out)
{
  int pix = blockIdx.x * 256 + threadIdx.x;
  if (pix >= P) return;
  out[pix] = kcF[pix] * d3[pix] + kcF[(size_t)P + pix] * d5[pix] +
             kcF[(size_t)2 * P + pix] * d7[pix];
}

} // namespace spn

// ------------------------------------------------------------------
extern "C" void kernel_launch(void* const* d_in, const int* in_sizes, int n_in,
                              void* d_out, int out_size, void* d_ws, size_t ws_size,
                              hipStream_t stream)
{
  using namespace spn;
  (void)in_sizes; (void)n_in; (void)out_size; (void)ws_size;

  // inputs in setup_inputs() dict order (params dict in insertion order;
  // each conv contributes .w .b .g .be .m .v)
  const float* feature1 = (const float*)d_in[0];
  const float* feature2 = (const float*)d_in[1];
  const float* depth    = (const float*)d_in[2];
  const float* sparse   = (const float*)d_in[3];
  const float* fbias    = (const float*)d_in[4];
  auto cp = [&](int conv, int which) -> const float* {
    return (const float*)d_in[5 + conv * 6 + which];
  };
  // conv idx: 0 proj1, 1 proj2, 2 mask, 3 kconf, 4 g3, 5 g5, 6 g7,
  //           7 mask_s2, 8 kconf_s2, 9 g3_s2, 10 g5_s2, 11 g7_s2
  // which:    0 w, 1 b, 2 g, 3 be, 4 m, 5 v

  // ---- workspace carve ----
  char* base = (char*)d_ws;
  size_t off = 0;
  auto carve = [&](size_t bytes) -> char* {
    char* p = base + off;
    off = (off + bytes + 255) & ~(size_t)255;
    return p;
  };
  _Float16* wpackF = (_Float16*)carve((size_t)9 * 6 * 512 * 2);
  _Float16* wpackH = (_Float16*)carve((size_t)18 * 6 * 512 * 2);
  float2* sbF = (float2*)carve(96 * sizeof(float2));
  float2* sbH = (float2*)carve(96 * sizeof(float2));
  float2* tmpBox = (float2*)carve((size_t)P * sizeof(float2));
  float* sparseF = (float*)carve((size_t)P * 4);
  float* validF  = (float*)carve((size_t)P * 4);
  float* ds2  = (float*)carve((size_t)P2 * 4);
  float* vms2 = (float*)carve((size_t)P2 * 4);
  _Float16* f1h = (_Float16*)carve((size_t)P * 32 * 2);
  _Float16* f2h = (_Float16*)carve((size_t)P2 * 64 * 2);
  float* maskF = (float*)carve((size_t)P * 4);
  float* kcF   = (float*)carve((size_t)3 * P * 4);
  float* maskS = (float*)carve((size_t)P2 * 4);
  float* kcS   = (float*)carve((size_t)3 * P2 * 4);
  _Float16* g3F = (_Float16*)carve((size_t)9  * P * 2);
  _Float16* g5F = (_Float16*)carve((size_t)25 * P * 2);
  _Float16* g7F = (_Float16*)carve((size_t)49 * P * 2);
  _Float16* g3S = (_Float16*)carve((size_t)9  * P2 * 2);
  _Float16* g5S = (_Float16*)carve((size_t)25 * P2 * 2);
  _Float16* g7S = (_Float16*)carve((size_t)49 * P2 * 2);
  float* dA = (float*)carve((size_t)3 * P * 4);
  float* dB = (float*)carve((size_t)3 * P * 4);
  float* refined = (float*)carve((size_t)P * 4);

  dim3 blk(256);

  // ---- weight/BN prep ----
  prep_w_kernel<<<(9 * 6 * 512 + 255) / 256, blk, 0, stream>>>(
      wpackF, 32, 9, cp(2, 0), cp(3, 0), cp(4, 0), cp(5, 0), cp(6, 0));
  prep_w_kernel<<<(18 * 6 * 512 + 255) / 256, blk, 0, stream>>>(
      wpackH, 64, 18, cp(7, 0), cp(8, 0), cp(9, 0), cp(10, 0), cp(11, 0));
  prep_sb_kernel<<<1, 128, 0, stream>>>(sbF,
      cp(2,1),cp(2,2),cp(2,3),cp(2,4),cp(2,5),
      cp(3,1),cp(3,2),cp(3,3),cp(3,4),cp(3,5),
      cp(4,1),cp(4,2),cp(4,3),cp(4,4),cp(4,5),
      cp(5,1),cp(5,2),cp(5,3),cp(5,4),cp(5,5),
      cp(6,1),cp(6,2),cp(6,3),cp(6,4),cp(6,5));
  prep_sb_kernel<<<1, 128, 0, stream>>>(sbH,
      cp(7,1),cp(7,2),cp(7,3),cp(7,4),cp(7,5),
      cp(8,1),cp(8,2),cp(8,3),cp(8,4),cp(8,5),
      cp(9,1),cp(9,2),cp(9,3),cp(9,4),cp(9,5),
      cp(10,1),cp(10,2),cp(10,3),cp(10,4),cp(10,5),
      cp(11,1),cp(11,2),cp(11,3),cp(11,4),cp(11,5));

  // ---- sparse preprocessing ----
  box_h_kernel<<<P / 256, blk, 0, stream>>>(sparse, tmpBox);
  box_v_filter_kernel<<<P / 256, blk, 0, stream>>>(sparse, depth, tmpBox, fbias,
                                                   sparseF, validF);
  downsample_kernel<<<P2 / 256, blk, 0, stream>>>(sparseF, validF, ds2, vms2);

  // ---- proj 1x1 convs (fp16 pixel-major outputs) ----
  proj_kernel<32><<<P / 256, blk, 0, stream>>>(feature1, cp(0, 0), cp(0, 1),
      cp(0, 2), cp(0, 3), cp(0, 4), cp(0, 5), f1h, P);
  proj_kernel<64><<<P2 / 256, blk, 0, stream>>>(feature2, cp(1, 0), cp(1, 1),
      cp(1, 2), cp(1, 3), cp(1, 4), cp(1, 5), f2h, P2);

  // ---- fused 3x3 conv heads via WMMA ----
  conv3_gemm_kernel<32, H, W><<<P / 64, blk, 0, stream>>>(
      f1h, wpackF, sbF, validF, maskF, kcF, g3F, g5F, g7F);
  conv3_gemm_kernel<64, H2, W2><<<P2 / 64, blk, 0, stream>>>(
      f2h, wpackH, sbH, vms2, maskS, kcS, g3S, g5S, g7S);

  // ---- s2 CSPN loop (6 iterations, dil=2, half-res guides) ----
  const float *c3 = depth, *c5 = depth, *c7 = depth;
  for (int it = 0; it < 6; ++it) {
    float* ob = (it & 1) ? dB : dA;
    cspn_s2_kernel<<<P / 256, blk, 0, stream>>>(
        c3, c5, c7, ob, ob + P, ob + 2 * (size_t)P, g3S, g5S, g7S, depth,
        maskS, ds2);
    c3 = ob; c5 = ob + P; c7 = ob + 2 * (size_t)P;
  }
  blend_s2_kernel<<<P / 256, blk, 0, stream>>>(c3, c5, c7, kcS, refined);

  // ---- full-res CSPN loop (6 iterations, dil=1) ----
  c3 = c5 = c7 = refined;
  for (int it = 0; it < 6; ++it) {
    float* ob = (it & 1) ? dB : dA;
    cspn_full_kernel<<<P / 256, blk, 0, stream>>>(
        c3, c5, c7, ob, ob + P, ob + 2 * (size_t)P, g3F, g5F, g7F, refined,
        maskF, sparseF);
    c3 = ob; c5 = ob + P; c7 = ob + 2 * (size_t)P;
  }
  blend_final_kernel<<<P / 256, blk, 0, stream>>>(c3, c5, c7, kcF,
                                                  (float*)d_out);
}
